// LocalPoolPointnet_20822001451315
// MI455X (gfx1250) — compile-verified, matrix operand order checked by reading, not run on hardware
//
#include <hip/hip_runtime.h>
#include <hip/hip_bf16.h>

#define RESO 32
#define GRID (RESO * RESO * RESO)
#define HID  32
#define CDIM 32
#define NBLK 5
#define BATCH 2

typedef __attribute__((ext_vector_type(16))) _Float16 v16h;
typedef __attribute__((ext_vector_type(8)))  float    v8f;

// ---- fragment index helpers -------------------------------------------------
// A/B fragment storage: [frag][lane 0..31][half 0..15] (1 KB per 16x32 tile).
// Within a 16x32 tile, element (row m, k kk) lives at:
//   lane = m + 16*((kk>>3)&1) ; half = ((kk>>4)<<3) + (kk&7)
static __device__ __forceinline__ int frag_pos(int m, int kk) {
    return (m + 16 * ((kk >> 3) & 1)) * 16 + ((kk >> 4) << 3) + (kk & 7);
}

static __device__ __forceinline__ v8f wmma_f16(v16h a, v16h b, v8f c) {
    return __builtin_amdgcn_wmma_f32_16x16x32_f16(
        false, a, false, b, (short)0, c, false, false);
}

static __device__ __forceinline__ v16h load_frag(const _Float16* __restrict__ buf,
                                                 int f, int lane) {
    return *((const v16h*)(buf + f * 512 + lane * 16));
}

static __device__ __forceinline__ v16h relu16(v16h a) {
    v16h r;
    const _Float16 z = (_Float16)0.0f;
#pragma unroll
    for (int i = 0; i < 16; ++i) r[i] = a[i] > z ? a[i] : z;
    return r;
}

// Weight fragment table (60 frags, f16, built once by k_prep):
//  f 0..7   : fc_pos  (kt = f>>2, nt = f&3)           K<=60 pad to 64, N=64
//  f 8..57  : block i: base = 8 + i*10
//             +0..3   w0  (kt*2+nt)   64x32
//             +4..5   w1  (nt)        32x32
//             +6..9   wsc (kt*2+nt)   64x32
//  f 58..59 : fc_c (nt)               32x32
#define WF_FP   0
#define WF_BLK(i) (8 + (i) * 10)
#define WF_FCC  58
#define WF_TOTAL 60

__global__ void k_prep(const float* __restrict__ fpw, const float* __restrict__ bw0,
                       const float* __restrict__ bw1, const float* __restrict__ bws,
                       const float* __restrict__ fcw, _Float16* __restrict__ wf)
{
    int t = blockIdx.x * blockDim.x + threadIdx.x;   // one thread = one (frag, lane)
    if (t >= WF_TOTAL * 32) return;
    int f = t >> 5, lane = t & 31;
    const float* w; int ldn, k0, n0, kmax;
    if (f < 8)       { w = fpw; ldn = 64; kmax = 60; k0 = (f >> 2) * 32; n0 = (f & 3) * 16; }
    else if (f < 58) {
        int i = (f - 8) / 10, j = (f - 8) % 10;
        if (j < 4)      { w = bw0 + i * 2048; ldn = 32; kmax = 64; k0 = (j >> 1) * 32; n0 = (j & 1) * 16; }
        else if (j < 6) { w = bw1 + i * 1024; ldn = 32; kmax = 32; k0 = 0;             n0 = (j - 4) * 16; }
        else            { int jj = j - 6;
                          w = bws + i * 2048; ldn = 32; kmax = 64; k0 = (jj >> 1) * 32; n0 = (jj & 1) * 16; }
    } else           { w = fcw; ldn = 32; kmax = 32; k0 = 0; n0 = (f - 58) * 16; }

    int n = lane & 15, hh = (lane >> 4) & 1;
    _Float16* dst = wf + f * 512 + lane * 16;
#pragma unroll
    for (int i = 0; i < 16; ++i) {
        int k = k0 + ((i >> 3) << 4) + (i & 7) + 8 * hh;
        float v = (k < kmax) ? w[k * ldn + n0 + n] : 0.0f;
        dst[i] = (_Float16)v;
    }
}

// ---- shared resblock: out = x@ws + relu(relu(relu(x)@w0+b0))@w1 + b1 --------
static __device__ __forceinline__ void resblock_wmma(
    const _Float16* xf,       // LDS, 2 frags (16x64 activation)
    _Float16*       hf,       // LDS, 1 frag
    const _Float16* __restrict__ wf, int fblk,
    const float* __restrict__ b0, const float* __restrict__ b1,
    _Float16* __restrict__ netf_out,   // global, this tile's frag
    int lane, int nvalid)
{
    v8f zero = {};
    v8f h0 = zero, h1 = zero;
#pragma unroll
    for (int kt = 0; kt < 2; ++kt) {
        v16h ar = relu16(load_frag(xf, kt, lane));
        h0 = wmma_f16(ar, load_frag(wf, fblk + kt * 2 + 0, lane), h0);
        h1 = wmma_f16(ar, load_frag(wf, fblk + kt * 2 + 1, lane), h1);
    }
    int n = lane & 15, mh = (lane >> 4) & 1;
#pragma unroll
    for (int r = 0; r < 8; ++r) {
        int m = mh * 8 + r;
        float v0 = h0[r] + b0[n];
        float v1 = h1[r] + b0[n + 16];
        hf[frag_pos(m, n)]      = (_Float16)(v0 > 0.0f ? v0 : 0.0f);
        hf[frag_pos(m, n + 16)] = (_Float16)(v1 > 0.0f ? v1 : 0.0f);
    }
    v8f o0 = zero, o1 = zero;
    v16h ah = load_frag(hf, 0, lane);
    o0 = wmma_f16(ah, load_frag(wf, fblk + 4, lane), o0);
    o1 = wmma_f16(ah, load_frag(wf, fblk + 5, lane), o1);
#pragma unroll
    for (int kt = 0; kt < 2; ++kt) {
        v16h ax = load_frag(xf, kt, lane);
        o0 = wmma_f16(ax, load_frag(wf, fblk + 6 + kt * 2 + 0, lane), o0);
        o1 = wmma_f16(ax, load_frag(wf, fblk + 6 + kt * 2 + 1, lane), o1);
    }
#pragma unroll
    for (int r = 0; r < 8; ++r) {
        int m = mh * 8 + r;
        if (m < nvalid) {
            netf_out[frag_pos(m, n)]      = (_Float16)(o0[r] + b1[n]);
            netf_out[frag_pos(m, n + 16)] = (_Float16)(o1[r] + b1[n + 16]);
        }
    }
}

// ---- kernel 1: pos-encode + fc_pos + resblock0 + grid index -----------------
__global__ void k_encode(const float* __restrict__ p,
                         const _Float16* __restrict__ wf,
                         const float* __restrict__ fpb,
                         const float* __restrict__ b0, const float* __restrict__ b1,
                         _Float16* __restrict__ netf, int* __restrict__ gidx,
                         int NP, int T)
{
    __shared__ __align__(32) _Float16 xls[8][1024];
    __shared__ __align__(32) _Float16 hls[8][512];
    int wave = threadIdx.x >> 5, lane = threadIdx.x & 31;
    int tile = blockIdx.x * 8 + wave;
    int base = tile * 16;
    if (base >= NP) return;                 // wave-uniform guard: EXEC stays full
    _Float16* xf = xls[wave];
    _Float16* hf = hls[wave];

    int m = lane & 15, hh = (lane >> 4) & 1;
    int pt = base + m; if (pt >= NP) pt = NP - 1;
    float px = p[pt * 3 + 0], py = p[pt * 3 + 1], pz = p[pt * 3 + 2];

    if (hh == 0 && (base + m) < NP) {
        const float inv = 1.0f / (1.0f + 0.1f + 1e-3f);
        float nx = fminf(fmaxf(px * inv + 0.5f, 0.0f), 1.0f - 1e-3f);
        float ny = fminf(fmaxf(py * inv + 0.5f, 0.0f), 1.0f - 1e-3f);
        float nz = fminf(fmaxf(pz * inv + 0.5f, 0.0f), 1.0f - 1e-3f);
        int ix = (int)floorf(nx * (float)RESO);
        int iy = (int)floorf(ny * (float)RESO);
        int iz = (int)floorf(nz * (float)RESO);
        int b  = (pt >= T) ? 1 : 0;         // BATCH == 2
        gidx[base + m] = ix + RESO * (iy + RESO * iz) + b * GRID;
    }

    // fc_pos: PE(60 -> pad 64) @ W ; PE built directly into A fragments.
    // Per element, k takes only two values (kbase, kbase+8) selected by hh.
    // All derived ints (l, rem, axis, phase, validity) are compile-time for each
    // variant, so codegen is: 2 FMA (angle) + 1 cndmask + 1 v_sin, branch-free.
    float qx = 2.0f * px - 1.0f, qy = 2.0f * py - 1.0f, qz = 2.0f * pz - 1.0f;
    v8f zero = {};
    v8f acc[4] = {zero, zero, zero, zero};
#pragma unroll
    for (int kt = 0; kt < 2; ++kt) {
        v16h a;
#pragma unroll
        for (int i = 0; i < 16; ++i) {
            const int kbase = kt * 32 + ((i >> 3) << 4) + (i & 7);  // literal after unroll
            float ang0, ang1;
            bool  val1;
            {
                const int k = kbase;                 // hh = 0 variant (always < 60 here)
                const int l = k / 6, rem = k % 6, ax = rem % 3;
                const float freq  = 3.14159265358979f * (float)(1 << l);
                const float phase = (rem >= 3) ? 1.57079632679f : 0.0f;  // cos = sin(+pi/2)
                const float q = (ax == 0) ? qx : (ax == 1 ? qy : qz);
                ang0 = q * freq + phase;
            }
            {
                const int k = kbase + 8;             // hh = 1 variant
                val1 = (k < 60);
                const int kk = val1 ? k : 0;
                const int l = kk / 6, rem = kk % 6, ax = rem % 3;
                const float freq  = 3.14159265358979f * (float)(1 << l);
                const float phase = (rem >= 3) ? 1.57079632679f : 0.0f;
                const float q = (ax == 0) ? qx : (ax == 1 ? qy : qz);
                ang1 = q * freq + phase;
            }
            float v;
            if (val1) {                              // compile-time condition
                v = __sinf(hh ? ang1 : ang0);
            } else {                                 // hh=1 element is zero padding
                float s = __sinf(ang0);
                v = hh ? 0.0f : s;
            }
            a[i] = (_Float16)v;
        }
#pragma unroll
        for (int nt = 0; nt < 4; ++nt)
            acc[nt] = wmma_f16(a, load_frag(wf, WF_FP + kt * 4 + nt, lane), acc[nt]);
    }
    int n = lane & 15;
#pragma unroll
    for (int nt = 0; nt < 4; ++nt)
#pragma unroll
        for (int r = 0; r < 8; ++r) {
            int k = nt * 16 + n, mm = hh * 8 + r;
            xf[(k >> 5) * 512 + frag_pos(mm, k & 31)] =
                (_Float16)(acc[nt][r] + fpb[k]);
        }

    int nvalid = NP - base; if (nvalid > 16) nvalid = 16;
    resblock_wmma(xf, hf, wf, WF_BLK(0), b0, b1, netf + (size_t)tile * 512, lane, nvalid);
}

// ---- pooling ----------------------------------------------------------------
__global__ void k_clear(unsigned* __restrict__ buf, int nelem) {
    int i = blockIdx.x * blockDim.x + threadIdx.x;
    if (i < nelem) buf[i] = 0u;
}

__global__ void k_scatter_max(const _Float16* __restrict__ netf, const int* __restrict__ gidx,
                              unsigned* __restrict__ cell, int NP)
{
    int i = blockIdx.x * blockDim.x + threadIdx.x;
    if (i >= NP * HID) return;
    int pt = i >> 5, c = i & (HID - 1);
    float v = (float)netf[(size_t)(pt >> 4) * 512 + frag_pos(pt & 15, c)];
    unsigned bits = __float_as_uint(v);
    unsigned s = (bits & 0x80000000u) ? ~bits : (bits | 0x80000000u);
    atomicMax(&cell[(size_t)gidx[pt] * HID + c], s);
}

// ---- kernels 2..5: gather pooled + concat + WMMA resblock -------------------
__global__ void k_block(_Float16* __restrict__ netf, const unsigned* __restrict__ cell,
                        const int* __restrict__ gidx,
                        const _Float16* __restrict__ wf, int fblk,
                        const float* __restrict__ b0, const float* __restrict__ b1,
                        int NP)
{
    __shared__ __align__(32) _Float16 xls[8][1024];
    __shared__ __align__(32) _Float16 hls[8][512];
    int wave = threadIdx.x >> 5, lane = threadIdx.x & 31;
    int tile = blockIdx.x * 8 + wave;
    int base = tile * 16;
    if (base >= NP) return;
    _Float16* xf = xls[wave];
    _Float16* hf = hls[wave];

    __builtin_prefetch(wf + fblk * 512, 0, 1);   // global_prefetch_b8

    // x tile 0 = this tile's net fragment: one vector copy per lane
    *((v16h*)(xf + lane * 16)) = *((const v16h*)(netf + (size_t)tile * 512 + lane * 16));

    // x tile 1 = pooled features (decode order-preserving uint -> f32 -> f16)
    for (int idx = lane; idx < 512; idx += 32) {
        int mm = idx >> 5, c = idx & 31;
        int pt = base + mm; if (pt >= NP) pt = NP - 1;
        unsigned s = cell[(size_t)gidx[pt] * HID + c];
        unsigned bits = (s & 0x80000000u) ? (s & 0x7FFFFFFFu) : ~s;
        xf[512 + frag_pos(mm, c)] = (_Float16)__uint_as_float(bits);
    }
    int nvalid = NP - base; if (nvalid > 16) nvalid = 16;
    resblock_wmma(xf, hf, wf, fblk, b0, b1, netf + (size_t)tile * 512, lane, nvalid);
}

// ---- fc_c + scatter-mean accumulation ---------------------------------------
__global__ void k_fcc(const _Float16* __restrict__ netf, const int* __restrict__ gidx,
                      const _Float16* __restrict__ wf, const float* __restrict__ fcb,
                      float* __restrict__ sums, float* __restrict__ cnt, int NP)
{
    int wave = threadIdx.x >> 5, lane = threadIdx.x & 31;
    int tile = blockIdx.x * 8 + wave;
    int base = tile * 16;
    if (base >= NP) return;

    v16h a = *((const v16h*)(netf + (size_t)tile * 512 + lane * 16));  // direct A frag
    v8f zero = {};
    v8f c0 = zero, c1 = zero;
    c0 = wmma_f16(a, load_frag(wf, WF_FCC + 0, lane), c0);
    c1 = wmma_f16(a, load_frag(wf, WF_FCC + 1, lane), c1);

    int n = lane & 15, mh = (lane >> 4) & 1;
#pragma unroll
    for (int r = 0; r < 8; ++r) {
        int pt = base + mh * 8 + r;
        if (pt < NP) {
            int g = gidx[pt];
            atomicAdd(&sums[(size_t)g * CDIM + n],      c0[r] + fcb[n]);
            atomicAdd(&sums[(size_t)g * CDIM + n + 16], c1[r] + fcb[n + 16]);
            if (n == 0) atomicAdd(&cnt[g], 1.0f);
        }
    }
}

// ---- mean + transpose to [B, CDIM, G] ---------------------------------------
__global__ void k_final(const float* __restrict__ sums, const float* __restrict__ cnt,
                        float* __restrict__ out, int total)
{
    int i = blockIdx.x * blockDim.x + threadIdx.x;
    if (i >= total) return;
    int g = i % GRID;
    int c = (i / GRID) % CDIM;
    int b = i / (GRID * CDIM);
    float nct = cnt[b * GRID + g];
    out[i] = sums[((size_t)(b * GRID + g)) * CDIM + c] / fmaxf(nct, 1.0f);
}

extern "C" void kernel_launch(void* const* d_in, const int* in_sizes, int n_in,
                              void* d_out, int out_size, void* d_ws, size_t ws_size,
                              hipStream_t stream)
{
    const float* p   = (const float*)d_in[0];
    const float* fpw = (const float*)d_in[1];
    const float* fpb = (const float*)d_in[2];
    const float* bw0 = (const float*)d_in[3];
    const float* bb0 = (const float*)d_in[4];
    const float* bw1 = (const float*)d_in[5];
    const float* bb1 = (const float*)d_in[6];
    const float* bws = (const float*)d_in[7];
    const float* fcw = (const float*)d_in[8];
    const float* fcb = (const float*)d_in[9];
    float* out = (float*)d_out;

    int NP = in_sizes[0] / 3;       // B*T points
    int T  = NP / BATCH;
    int BG = BATCH * GRID;
    int ntiles = (NP + 15) / 16;

    char* ws = (char*)d_ws;
    _Float16* wfrag = (_Float16*)ws;  ws += (size_t)WF_TOTAL * 512 * sizeof(_Float16);
    _Float16* netf  = (_Float16*)ws;  ws += (size_t)ntiles * 512 * sizeof(_Float16);
    int*      gidx  = (int*)ws;       ws += (size_t)NP * sizeof(int);
    unsigned* cell  = (unsigned*)ws;  ws += (size_t)BG * HID * sizeof(unsigned);
    float*    sums  = (float*)ws;     ws += (size_t)BG * CDIM * sizeof(float);
    float*    cnt   = (float*)ws;

    int nblocks = (ntiles + 7) / 8;  // 8 wave-tiles per 256-thread block

    k_prep<<<(WF_TOTAL * 32 + 255) / 256, 256, 0, stream>>>(fpw, bw0, bw1, bws, fcw, wfrag);

    k_encode<<<nblocks, 256, 0, stream>>>(p, wfrag, fpb, bb0, bb1, netf, gidx, NP, T);

    for (int i = 1; i < NBLK; ++i) {
        int ce = BG * HID;
        k_clear<<<(ce + 255) / 256, 256, 0, stream>>>(cell, ce);
        k_scatter_max<<<(NP * HID + 255) / 256, 256, 0, stream>>>(netf, gidx, cell, NP);
        k_block<<<nblocks, 256, 0, stream>>>(netf, cell, gidx, wfrag, WF_BLK(i),
                                             bb0 + i * HID, bb1 + i * HID, NP);
    }

    k_clear<<<(BG * CDIM + 255) / 256, 256, 0, stream>>>((unsigned*)sums, BG * CDIM);
    k_clear<<<(BG + 255) / 256, 256, 0, stream>>>((unsigned*)cnt, BG);
    k_fcc<<<nblocks, 256, 0, stream>>>(netf, gidx, wfrag, fcb, sums, cnt, NP);

    int total = BATCH * CDIM * GRID;
    k_final<<<(total + 255) / 256, 256, 0, stream>>>(sums, cnt, out, total);
}
